// EncoderClassicGAT2_3917010174727
// MI455X (gfx1250) — matrix operand order repaired and checked by reading, not compiled, not run on hardware
//
#include <hip/hip_runtime.h>

// ---------------------------------------------------------------------------
// Problem constants (from reference)
// ---------------------------------------------------------------------------
#define NSN 20000
#define NAN_ 20000
#define EAE 320000
#define ESE 640000
#define HID 64

// K paddings (multiples of 32 for 16x16x32 bf16 WMMA k-steps)
#define KP_A2S 96    // 5 + 16 + 64 = 85 -> 96
#define KP_S2S 160   // 5 + 128    = 133 -> 160
#define KP_NODE 224  // 2 + 192    = 194 -> 224

typedef __bf16 bf16_t;
typedef bf16_t v16bf __attribute__((ext_vector_type(16)));
typedef float  v8f   __attribute__((ext_vector_type(8)));

// ---------------------------------------------------------------------------
// Fast math: use CDNA5 hardware transcendentals, branch-free fallbacks.
// ---------------------------------------------------------------------------
__device__ __forceinline__ float fast_rcp(float x) {
#if __has_builtin(__builtin_amdgcn_rcpf)
  return __builtin_amdgcn_rcpf(x);        // v_rcp_f32
#else
  return 1.0f / x;
#endif
}

__device__ __forceinline__ float fast_exp2(float x) {
#if __has_builtin(__builtin_amdgcn_exp2f)
  return __builtin_amdgcn_exp2f(x);       // v_exp_f32
#else
  return exp2f(x);
#endif
}

// exp(x) for x <= 0 (softmax: logit - max); underflow->0 is the right limit.
__device__ __forceinline__ float fast_exp(float x) {
  return fast_exp2(x * 1.4426950408889634f);
}

__device__ __forceinline__ float fast_tanh(float x) {
#if __has_builtin(__builtin_amdgcn_tanhf)
  return __builtin_amdgcn_tanhf(x);       // v_tanh_f32 (gfx1250 trans op)
#elif __has_builtin(__builtin_amdgcn_tanh_f32)
  return __builtin_amdgcn_tanh_f32(x);
#else
  // branch-free: tanh(x) = 1 - 2/(e^{2x}+1); saturates correctly at +-inf
  float e2 = fast_exp2(2.8853900817779268f * x);  // e^{2x}
  return 1.0f - 2.0f * fast_rcp(e2 + 1.0f);
#endif
}

// ---------------------------------------------------------------------------
// Helpers
// ---------------------------------------------------------------------------
__device__ __forceinline__ bf16_t f2bf(float f) {
  unsigned u = __float_as_uint(f);
  u += 0x7FFFu + ((u >> 16) & 1u);          // round-to-nearest-even
  unsigned short s = (unsigned short)(u >> 16);
  return __builtin_bit_cast(bf16_t, s);
}

// monotone float <-> uint encoding for atomicMax on floats
__device__ __forceinline__ unsigned encf(float f) {
  unsigned u = __float_as_uint(f);
  return (u & 0x80000000u) ? ~u : (u | 0x80000000u);
}
__device__ __forceinline__ float decf(unsigned u) {
  unsigned b = (u & 0x80000000u) ? (u & 0x7FFFFFFFu) : ~u;
  return __uint_as_float(b);
}
#define ENC_NEG_INF 0x007FFFFFu   // encf(-inf)

// wave-local LDS producer->consumer ordering (single wave owns its buffers,
// so a full barrier is unnecessary; just drain DScnt and stop reordering)
__device__ __forceinline__ void lds_wave_sync() {
  asm volatile("s_wait_dscnt 0x0" ::: "memory");
}

struct U32x8 { uint4 a, b; };

// Load a 16-half fragment as two aligned b128 loads.
//   A-frag (16-bit A 16x32 ISA layout): halves 0..7 at p, 8..15 at p+16
//   B-frag (lane = column, same K order as A): halves 0..7 at p, 8..15 at p+16
__device__ __forceinline__ v16bf ld16(const bf16_t* p, int hi_off_halfs) {
  U32x8 r;
  r.a = *(const uint4*)(p);
  r.b = *(const uint4*)(p + hi_off_halfs);
  return __builtin_bit_cast(v16bf, r);
}

__device__ __forceinline__ v16bf load_a_frag(const bf16_t* base, int ldk,
                                             int kbase, int lane) {
  const int row = lane & 15, g = lane >> 4;
  return ld16(base + row * ldk + kbase + 8 * g, 16);
}
__device__ __forceinline__ v16bf load_b_frag(const bf16_t* wt, int ldk, int t,
                                             int kbase, int lane) {
  const int col = lane & 15, g = lane >> 4;
  return ld16(wt + (t * 16 + col) * ldk + kbase + 8 * g, 16);
}

__device__ __forceinline__ v8f wmma_bf16(v16bf a, v16bf b, v8f c) {
  return __builtin_amdgcn_wmma_f32_16x16x32_bf16(false, a, false, b, (short)0,
                                                 c, false, false);
}

// ---------------------------------------------------------------------------
// 3-layer MLP on a 16-row tile:  tanh(tanh(X W1 + b1) W2 + b2) W3 + b3
//   xin : [16][KPAD] bf16 in LDS (row-major)
//   w*  : transposed bf16 weights [64][K] in global (L2/WGP$-resident)
//   s2  : [16][64] bf16 LDS scratch (per wave)
//   out : out[t][v] is D element (M = v + 8*(lane>>4), N = 16t + (lane&15))
// ---------------------------------------------------------------------------
template <int KPAD>
__device__ __forceinline__ void mlp3(const bf16_t* __restrict__ xin,
                                     const bf16_t* __restrict__ w1,
                                     const float* __restrict__ b1,
                                     const bf16_t* __restrict__ w2,
                                     const float* __restrict__ b2,
                                     const bf16_t* __restrict__ w3,
                                     const float* __restrict__ b3,
                                     bf16_t* __restrict__ s2, int lane,
                                     float out[4][8]) {
  constexpr int KS = KPAD / 32;
  const int col = lane & 15, gHi = lane >> 4;

  v16bf a[KS];
#pragma unroll
  for (int s = 0; s < KS; ++s) a[s] = load_a_frag(xin, KPAD, 32 * s, lane);

  // ---- layer 1: 16xKPAD @ KPADx64 ----
#pragma unroll
  for (int t = 0; t < 4; ++t) {
    v8f c = {};
#pragma unroll
    for (int s = 0; s < KS; ++s)
      c = wmma_bf16(a[s], load_b_frag(w1, KPAD, t, 32 * s, lane), c);
    float bias = b1[t * 16 + col];
#pragma unroll
    for (int v = 0; v < 8; ++v) {
      int M = v + 8 * gHi;
      s2[M * 64 + t * 16 + col] = f2bf(fast_tanh(c[v] + bias));
    }
  }
  lds_wave_sync();

  // ---- layer 2: 16x64 @ 64x64 ----
  v16bf a2[2];
#pragma unroll
  for (int s = 0; s < 2; ++s) a2[s] = load_a_frag(s2, 64, 32 * s, lane);
  lds_wave_sync();
#pragma unroll
  for (int t = 0; t < 4; ++t) {
    v8f c = {};
#pragma unroll
    for (int s = 0; s < 2; ++s)
      c = wmma_bf16(a2[s], load_b_frag(w2, 64, t, 32 * s, lane), c);
    float bias = b2[t * 16 + col];
#pragma unroll
    for (int v = 0; v < 8; ++v) {
      int M = v + 8 * gHi;
      s2[M * 64 + t * 16 + col] = f2bf(fast_tanh(c[v] + bias));
    }
  }
  lds_wave_sync();

  // ---- layer 3 (no activation) ----
#pragma unroll
  for (int s = 0; s < 2; ++s) a2[s] = load_a_frag(s2, 64, 32 * s, lane);
  lds_wave_sync();
#pragma unroll
  for (int t = 0; t < 4; ++t) {
    v8f c = {};
#pragma unroll
    for (int s = 0; s < 2; ++s)
      c = wmma_bf16(a2[s], load_b_frag(w3, 64, t, 32 * s, lane), c);
    float bias = b3[t * 16 + col];
#pragma unroll
    for (int v = 0; v < 8; ++v) out[t][v] = c[v] + bias;
  }
}

// ---------------------------------------------------------------------------
// Scatter epilogues (dst loaded once per row; atomics land in L2)
// ---------------------------------------------------------------------------
__device__ __forceinline__ void epilogue_max(const float outL[4][8],
                                             const int* __restrict__ edst,
                                             int ebase, unsigned* __restrict__ mx,
                                             int lane) {
  const int col = lane & 15, gHi = lane >> 4;
#pragma unroll
  for (int v = 0; v < 8; ++v) {
    const int dst = edst[ebase + v + 8 * gHi];
    unsigned* mrow = mx + dst * 64 + col;
#pragma unroll
    for (int t = 0; t < 4; ++t) atomicMax(mrow + t * 16, encf(outL[t][v]));
  }
}

__device__ __forceinline__ void epilogue_expsum(
    const float outL[4][8], const float outM[4][8],
    const int* __restrict__ edst, int ebase, const unsigned* __restrict__ mx,
    float* __restrict__ den, float* __restrict__ acc, int lane) {
  const int col = lane & 15, gHi = lane >> 4;
#pragma unroll
  for (int v = 0; v < 8; ++v) {
    const int base = edst[ebase + v + 8 * gHi] * 64 + col;
#pragma unroll
    for (int t = 0; t < 4; ++t) {
      const int idx = base + t * 16;
      const float ee = fast_exp(outL[t][v] - decf(mx[idx]));
      unsafeAtomicAdd(den + idx, ee);
      unsafeAtomicAdd(acc + idx, ee * outM[t][v]);
    }
  }
}

// ---------------------------------------------------------------------------
// Workspace init
// ---------------------------------------------------------------------------
__global__ void init_ws_kernel(unsigned* mxa, unsigned* mxs, float* dena,
                               float* acca, float* dens, float* accs, int n) {
  int i = blockIdx.x * blockDim.x + threadIdx.x;
  if (i < n) {
    mxa[i] = ENC_NEG_INF;
    mxs[i] = ENC_NEG_INF;
    dena[i] = 0.f; acca[i] = 0.f; dens[i] = 0.f; accs[i] = 0.f;
  }
}

// Convert W[k][64] fp32 -> transposed padded bf16 [64][KP]
__global__ void prep_w_kernel(const float* __restrict__ W, int din, int KP,
                              bf16_t* __restrict__ out) {
  int i = blockIdx.x * blockDim.x + threadIdx.x;
  if (i < 64 * KP) {
    int n = i / KP, k = i - n * KP;
    out[i] = f2bf(k < din ? W[k * 64 + n] : 0.0f);
  }
}

// ---------------------------------------------------------------------------
// a2s edge kernel: features [pa[src], ps[dst], dis, u[src], h[dst]] (85 -> 96)
//   PASS1: logit MLP only, atomicMax of logits per (dst, channel)
//   PASS2: recompute logit + msg MLP, accumulate exp-sum and exp*msg
// ---------------------------------------------------------------------------
template <bool PASS1>
__global__ __launch_bounds__(256) void a2s_edge_kernel(
    const float* __restrict__ h, const float* __restrict__ u,
    const float* __restrict__ ps, const float* __restrict__ pa,
    const float* __restrict__ dis, const int* __restrict__ esrc,
    const int* __restrict__ edst, const bf16_t* wl1, const float* bl1,
    const bf16_t* wl2, const float* bl2, const bf16_t* wl3, const float* bl3,
    const bf16_t* wm1, const float* bm1, const bf16_t* wm2, const float* bm2,
    const bf16_t* wm3, const float* bm3, unsigned* __restrict__ mx,
    float* __restrict__ den, float* __restrict__ acc, int ntiles) {
  constexpr int KP = KP_A2S;
  constexpr int WPB = 8;
  __shared__ bf16_t stageA[WPB][16 * KP];
  __shared__ bf16_t stage2[WPB][16 * 64];

  const int wave = threadIdx.x >> 5, lane = threadIdx.x & 31;
  const int tile = blockIdx.x * WPB + wave;
  if (tile >= ntiles) return;
  const int ebase = tile * 16;
  bf16_t* sA = stageA[wave];
  bf16_t* s2 = stage2[wave];

  // gather: lane group g covers features [48g, 48g+48)
  {
    const int r = lane & 15, g = lane >> 4;
    const int e = ebase + r;
    const int src = esrc[e], dst = edst[e];
    bf16_t* srow = sA + r * KP;
    if (g == 0) {
      srow[0] = f2bf(pa[src * 2 + 0]);
      srow[1] = f2bf(pa[src * 2 + 1]);
      srow[2] = f2bf(ps[dst * 2 + 0]);
      srow[3] = f2bf(ps[dst * 2 + 1]);
      srow[4] = f2bf(dis[e]);
#pragma unroll
      for (int i = 0; i < 16; ++i) srow[5 + i] = f2bf(u[src * 16 + i]);
#pragma unroll
      for (int i = 0; i < 27; ++i) srow[21 + i] = f2bf(h[dst * 64 + i]);
    } else {
#pragma unroll
      for (int i = 0; i < 37; ++i) srow[48 + i] = f2bf(h[dst * 64 + 27 + i]);
#pragma unroll
      for (int i = 0; i < 11; ++i) srow[85 + i] = f2bf(0.0f);
    }
  }
  lds_wave_sync();

  float outL[4][8], outM[4][8];
  if constexpr (!PASS1)
    mlp3<KP>(sA, wm1, bm1, wm2, bm2, wm3, bm3, s2, lane, outM);
  mlp3<KP>(sA, wl1, bl1, wl2, bl2, wl3, bl3, s2, lane, outL);

  if constexpr (PASS1)
    epilogue_max(outL, edst, ebase, mx, lane);
  else
    epilogue_expsum(outL, outM, edst, ebase, mx, den, acc, lane);
}

// ---------------------------------------------------------------------------
// s2s edge kernel: features [ps[src], ps[dst], dis, h[src], h[dst]] (133->160)
// ---------------------------------------------------------------------------
template <bool PASS1>
__global__ __launch_bounds__(256) void s2s_edge_kernel(
    const float* __restrict__ h, const float* __restrict__ ps,
    const float* __restrict__ dis, const int* __restrict__ esrc,
    const int* __restrict__ edst, const bf16_t* wl1, const float* bl1,
    const bf16_t* wl2, const float* bl2, const bf16_t* wl3, const float* bl3,
    const bf16_t* wm1, const float* bm1, const bf16_t* wm2, const float* bm2,
    const bf16_t* wm3, const float* bm3, unsigned* __restrict__ mx,
    float* __restrict__ den, float* __restrict__ acc, int ntiles) {
  constexpr int KP = KP_S2S;
  constexpr int WPB = 8;
  __shared__ bf16_t stageA[WPB][16 * KP];
  __shared__ bf16_t stage2[WPB][16 * 64];

  const int wave = threadIdx.x >> 5, lane = threadIdx.x & 31;
  const int tile = blockIdx.x * WPB + wave;
  if (tile >= ntiles) return;
  const int ebase = tile * 16;
  bf16_t* sA = stageA[wave];
  bf16_t* s2 = stage2[wave];

  {
    const int r = lane & 15, g = lane >> 4;
    const int e = ebase + r;
    const int src = esrc[e], dst = edst[e];
    bf16_t* srow = sA + r * KP;
    if (g == 0) {  // f 0..79
      srow[0] = f2bf(ps[src * 2 + 0]);
      srow[1] = f2bf(ps[src * 2 + 1]);
      srow[2] = f2bf(ps[dst * 2 + 0]);
      srow[3] = f2bf(ps[dst * 2 + 1]);
      srow[4] = f2bf(dis[e]);
#pragma unroll
      for (int i = 0; i < 64; ++i) srow[5 + i] = f2bf(h[src * 64 + i]);
#pragma unroll
      for (int i = 0; i < 11; ++i) srow[69 + i] = f2bf(h[dst * 64 + i]);
    } else {  // f 80..159
#pragma unroll
      for (int i = 0; i < 53; ++i) srow[80 + i] = f2bf(h[dst * 64 + 11 + i]);
#pragma unroll
      for (int i = 0; i < 27; ++i) srow[133 + i] = f2bf(0.0f);
    }
  }
  lds_wave_sync();

  float outL[4][8], outM[4][8];
  if constexpr (!PASS1)
    mlp3<KP>(sA, wm1, bm1, wm2, bm2, wm3, bm3, s2, lane, outM);
  mlp3<KP>(sA, wl1, bl1, wl2, bl2, wl3, bl3, s2, lane, outL);

  if constexpr (PASS1)
    epilogue_max(outL, edst, ebase, mx, lane);
  else
    epilogue_expsum(outL, outM, edst, ebase, mx, den, acc, lane);
}

// ---------------------------------------------------------------------------
// node updater: features [ps, h, sum_u, sum_h] (194 -> 224), write d_out fp32
// ---------------------------------------------------------------------------
__global__ __launch_bounds__(128) void node_kernel(
    const float* __restrict__ h, const float* __restrict__ ps,
    const float* __restrict__ dena, const float* __restrict__ acca,
    const float* __restrict__ dens, const float* __restrict__ accs,
    const bf16_t* w1, const float* b1, const bf16_t* w2, const float* b2,
    const bf16_t* w3, const float* b3, float* __restrict__ out, int ntiles) {
  constexpr int KP = KP_NODE;
  constexpr int WPB = 4;
  __shared__ bf16_t stageA[WPB][16 * KP];
  __shared__ bf16_t stage2[WPB][16 * 64];

  const int wave = threadIdx.x >> 5, lane = threadIdx.x & 31;
  const int tile = blockIdx.x * WPB + wave;
  if (tile >= ntiles) return;
  const int nbase = tile * 16;
  bf16_t* sA = stageA[wave];
  bf16_t* s2 = stage2[wave];

  {
    const int r = lane & 15, g = lane >> 4;
    const int n = nbase + r;
    bf16_t* srow = sA + r * KP;
    if (g == 0) {  // f 0..111
      srow[0] = f2bf(ps[n * 2 + 0]);
      srow[1] = f2bf(ps[n * 2 + 1]);
#pragma unroll
      for (int i = 0; i < 64; ++i) srow[2 + i] = f2bf(h[n * 64 + i]);
#pragma unroll
      for (int i = 0; i < 46; ++i) {
        const float d = dena[n * 64 + i];
        srow[66 + i] = f2bf(d > 0.f ? acca[n * 64 + i] * fast_rcp(d) : 0.f);
      }
    } else {  // f 112..223
#pragma unroll
      for (int i = 0; i < 18; ++i) {
        const float d = dena[n * 64 + 46 + i];
        srow[112 + i] = f2bf(d > 0.f ? acca[n * 64 + 46 + i] * fast_rcp(d) : 0.f);
      }
#pragma unroll
      for (int i = 0; i < 64; ++i) {
        const float d = dens[n * 64 + i];
        srow[130 + i] = f2bf(d > 0.f ? accs[n * 64 + i] * fast_rcp(d) : 0.f);
      }
#pragma unroll
      for (int i = 0; i < 30; ++i) srow[194 + i] = f2bf(0.0f);
    }
  }
  lds_wave_sync();

  float o[4][8];
  mlp3<KP>(sA, w1, b1, w2, b2, w3, b3, s2, lane, o);

  const int col = lane & 15, gHi = lane >> 4;
#pragma unroll
  for (int v = 0; v < 8; ++v) {
    const int n = nbase + v + 8 * gHi;
#pragma unroll
    for (int t = 0; t < 4; ++t) out[n * 64 + t * 16 + col] = o[t][v];
  }
}

// ---------------------------------------------------------------------------
// Host launcher
// ---------------------------------------------------------------------------
extern "C" void kernel_launch(void* const* d_in, const int* in_sizes, int n_in,
                              void* d_out, int out_size, void* d_ws,
                              size_t ws_size, hipStream_t stream) {
  (void)in_sizes; (void)n_in; (void)out_size; (void)ws_size;
  const float* h      = (const float*)d_in[0];
  const float* u      = (const float*)d_in[1];
  const float* ps     = (const float*)d_in[2];
  const float* pa     = (const float*)d_in[3];
  const float* dis_a  = (const float*)d_in[4];
  const float* dis_s  = (const float*)d_in[5];
  const int*   a_src  = (const int*)d_in[6];
  const int*   a_dst  = (const int*)d_in[7];
  const int*   s_src  = (const int*)d_in[8];
  const int*   s_dst  = (const int*)d_in[9];
  // Five MLP param tuples, 6 leaves each (W1,b1,W2,b2,W3,b3), flattened:
  const float* P[30];
  for (int i = 0; i < 30; ++i) P[i] = (const float*)d_in[10 + i];
  const float** u2h_l = P + 0;   // a2s logit
  const float** u2h_u = P + 6;   // a2s msg
  const float** h2h_l = P + 12;  // s2s logit
  const float** h2h_h = P + 18;  // s2s msg
  const float** updp  = P + 24;  // node updater

  // ---- workspace layout (~31 MB + 176 KB) ----
  const int NS64 = NSN * 64;
  unsigned* max_a = (unsigned*)d_ws;
  unsigned* max_s = max_a + NS64;
  float* den_a = (float*)(max_s + NS64);
  float* acc_a = den_a + NS64;
  float* den_s = acc_a + NS64;
  float* acc_s = den_s + NS64;
  bf16_t* wb = (bf16_t*)(acc_s + NS64);
  bf16_t* wAL1 = wb;                 wb += 64 * KP_A2S;
  bf16_t* wAL2 = wb;                 wb += 64 * 64;
  bf16_t* wAL3 = wb;                 wb += 64 * 64;
  bf16_t* wAM1 = wb;                 wb += 64 * KP_A2S;
  bf16_t* wAM2 = wb;                 wb += 64 * 64;
  bf16_t* wAM3 = wb;                 wb += 64 * 64;
  bf16_t* wSL1 = wb;                 wb += 64 * KP_S2S;
  bf16_t* wSL2 = wb;                 wb += 64 * 64;
  bf16_t* wSL3 = wb;                 wb += 64 * 64;
  bf16_t* wSM1 = wb;                 wb += 64 * KP_S2S;
  bf16_t* wSM2 = wb;                 wb += 64 * 64;
  bf16_t* wSM3 = wb;                 wb += 64 * 64;
  bf16_t* wN1  = wb;                 wb += 64 * KP_NODE;
  bf16_t* wN2  = wb;                 wb += 64 * 64;
  bf16_t* wN3  = wb;                 wb += 64 * 64;

  init_ws_kernel<<<(NS64 + 255) / 256, 256, 0, stream>>>(
      max_a, max_s, den_a, acc_a, den_s, acc_s, NS64);

  auto prep = [&](const float* W, int din, int KP, bf16_t* o) {
    prep_w_kernel<<<(64 * KP + 255) / 256, 256, 0, stream>>>(W, din, KP, o);
  };
  prep(u2h_l[0], 85, KP_A2S, wAL1); prep(u2h_l[2], 64, 64, wAL2); prep(u2h_l[4], 64, 64, wAL3);
  prep(u2h_u[0], 85, KP_A2S, wAM1); prep(u2h_u[2], 64, 64, wAM2); prep(u2h_u[4], 64, 64, wAM3);
  prep(h2h_l[0], 133, KP_S2S, wSL1); prep(h2h_l[2], 64, 64, wSL2); prep(h2h_l[4], 64, 64, wSL3);
  prep(h2h_h[0], 133, KP_S2S, wSM1); prep(h2h_h[2], 64, 64, wSM2); prep(h2h_h[4], 64, 64, wSM3);
  prep(updp[0], 194, KP_NODE, wN1);  prep(updp[2], 64, 64, wN2);  prep(updp[4], 64, 64, wN3);

  const int tilesA = EAE / 16, tilesS = ESE / 16, tilesN = (NSN + 15) / 16;

  // pass 1: per-(dst,channel) logit max
  a2s_edge_kernel<true><<<(tilesA + 7) / 8, 256, 0, stream>>>(
      h, u, ps, pa, dis_a, a_src, a_dst,
      wAL1, u2h_l[1], wAL2, u2h_l[3], wAL3, u2h_l[5],
      nullptr, nullptr, nullptr, nullptr, nullptr, nullptr,
      max_a, den_a, acc_a, tilesA);
  s2s_edge_kernel<true><<<(tilesS + 7) / 8, 256, 0, stream>>>(
      h, ps, dis_s, s_src, s_dst,
      wSL1, h2h_l[1], wSL2, h2h_l[3], wSL3, h2h_l[5],
      nullptr, nullptr, nullptr, nullptr, nullptr, nullptr,
      max_s, den_s, acc_s, tilesS);

  // pass 2: recompute logit + msg, accumulate exp-sum and exp*msg
  a2s_edge_kernel<false><<<(tilesA + 7) / 8, 256, 0, stream>>>(
      h, u, ps, pa, dis_a, a_src, a_dst,
      wAL1, u2h_l[1], wAL2, u2h_l[3], wAL3, u2h_l[5],
      wAM1, u2h_u[1], wAM2, u2h_u[3], wAM3, u2h_u[5],
      max_a, den_a, acc_a, tilesA);
  s2s_edge_kernel<false><<<(tilesS + 7) / 8, 256, 0, stream>>>(
      h, ps, dis_s, s_src, s_dst,
      wSL1, h2h_l[1], wSL2, h2h_l[3], wSL3, h2h_l[5],
      wSM1, h2h_h[1], wSM2, h2h_h[3], wSM3, h2h_h[5],
      max_s, den_s, acc_s, tilesS);

  // node update
  node_kernel<<<(tilesN + 3) / 4, 128, 0, stream>>>(
      h, ps, den_a, acc_a, den_s, acc_s,
      wN1, updp[1], wN2, updp[3], wN3, updp[5], (float*)d_out, tilesN);
}
